// SCAtt_3384434229774
// MI455X (gfx1250) — matrix operand order repaired and checked by reading, not compiled, not run on hardware
//
#include <hip/hip_runtime.h>

// ---------------------------------------------------------------------------
// SCAtt (dual attention) for MI455X / gfx1250.
// All contractions via v_wmma_f32_16x16x32_bf16 (bf16 operands, f32 accum).
// Every operand stored so WMMA fragments are contiguous b128 loads from L2.
// Wave tile 32x64 (2x4 frags), block tile 128x128 (8 waves, 4x2).
// ---------------------------------------------------------------------------

typedef __bf16 bf16;
typedef __attribute__((ext_vector_type(16))) __bf16 bf16x16;
typedef __attribute__((ext_vector_type(8)))  __bf16 bf16x8;
typedef __attribute__((ext_vector_type(8)))  float  f32x8;

static __device__ __forceinline__ f32x8 wmma_bf16(bf16x16 a, bf16x16 b, f32x8 c) {
  return __builtin_amdgcn_wmma_f32_16x16x32_bf16(
      /*neg_a=*/false, a, /*neg_b=*/false, b,
      /*c_mod=*/(short)0, c, /*reuse_a=*/false, /*reuse_b=*/false);
}

// A-matrix 16x32 bf16 fragment. Lane L: row = row0 + (L&15); hi = L>>4.
// elems 0..7 hold K = hi*8..hi*8+7, elems 8..15 hold K = 16+hi*8..23+hi*8.
static __device__ __forceinline__ bf16x16
ld_frag_a(const bf16* __restrict__ A, long row, int lda, int kc, int hi) {
  const bf16* p = A + row * (long)lda + kc;
  bf16x16 r;
  reinterpret_cast<bf16x8*>(&r)[0] = *reinterpret_cast<const bf16x8*>(p + hi * 8);
  reinterpret_cast<bf16x8*>(&r)[1] = *reinterpret_cast<const bf16x8*>(p + 16 + hi * 8);
  return r;
}

// B-matrix 32x16 bf16 fragment from transposed storage BT[col][k].
// Lane L: col = col0 + (L&15); elems 0..7 hold K = hi*8..hi*8+7, elems 8..15 hold K = 16+hi*8..23+hi*8.
static __device__ __forceinline__ bf16x16
ld_frag_b_row(const bf16* __restrict__ rowp, int kc, int hi) {
  const bf16* p = rowp + kc + hi * 8;
  bf16x16 r;
  reinterpret_cast<bf16x8*>(&r)[0] = reinterpret_cast<const bf16x8*>(p)[0];
  reinterpret_cast<bf16x8*>(&r)[1] = *reinterpret_cast<const bf16x8*>(p + 16);
  return r;
}

// ---------------------------------------------------------------------------
// Generic NT GEMM: out[M,N] = A[M,K] * B[K,N], B stored transposed BT[N,K].
// Block = 256 threads = 8 waves (4m x 2n), wave tile 32x64, block 128x128.
// epi==0: f32 natural out (+bias).  epi==1: bf16 transposed out * scale.
// ---------------------------------------------------------------------------
__global__ __launch_bounds__(256) void gemm_nt_kernel(
    int K,
    const bf16* __restrict__ A,  int lda, long bsA,
    const bf16* __restrict__ BT, int ldb, long bsBT,
    float* outF, bf16* outT, int ldo, long bsOut,
    const float* __restrict__ scale_ptr,
    const float* __restrict__ bias,
    int epi) {
  const int tid  = threadIdx.x;
  const int lane = tid & 31;
  const int wv   = tid >> 5;
  const int lo   = lane & 15;
  const int hi   = lane >> 4;
  const int mw   = wv >> 1;   // 0..3
  const int nw   = wv & 1;    // 0..1
  const int bz   = blockIdx.z;

  A  += (long)bz * bsA;
  BT += (long)bz * bsBT;
  if (outF) outF += (long)bz * bsOut;
  if (outT) outT += (long)bz * bsOut;

  const long row0 = (long)blockIdx.x * 128 + mw * 32;
  const long col0 = (long)blockIdx.y * 128 + nw * 64;

  f32x8 acc[2][4];
#pragma unroll
  for (int mi = 0; mi < 2; ++mi)
#pragma unroll
    for (int ni = 0; ni < 4; ++ni)
#pragma unroll
      for (int e = 0; e < 8; ++e) acc[mi][ni][e] = 0.0f;

  const bf16* br[4];
#pragma unroll
  for (int ni = 0; ni < 4; ++ni)
    br[ni] = BT + (col0 + ni * 16 + lo) * (long)ldb;

  for (int kc = 0; kc < K; kc += 32) {
    bf16x16 a0 = ld_frag_a(A, row0 + lo,      lda, kc, hi);
    bf16x16 a1 = ld_frag_a(A, row0 + 16 + lo, lda, kc, hi);
    bf16x16 b0 = ld_frag_b_row(br[0], kc, hi);
    bf16x16 b1 = ld_frag_b_row(br[1], kc, hi);
    bf16x16 b2 = ld_frag_b_row(br[2], kc, hi);
    bf16x16 b3 = ld_frag_b_row(br[3], kc, hi);
    acc[0][0] = wmma_bf16(a0, b0, acc[0][0]);
    acc[0][1] = wmma_bf16(a0, b1, acc[0][1]);
    acc[0][2] = wmma_bf16(a0, b2, acc[0][2]);
    acc[0][3] = wmma_bf16(a0, b3, acc[0][3]);
    acc[1][0] = wmma_bf16(a1, b0, acc[1][0]);
    acc[1][1] = wmma_bf16(a1, b1, acc[1][1]);
    acc[1][2] = wmma_bf16(a1, b2, acc[1][2]);
    acc[1][3] = wmma_bf16(a1, b3, acc[1][3]);
  }

  const float scale = scale_ptr ? scale_ptr[0] : 1.0f;
#pragma unroll
  for (int mi = 0; mi < 2; ++mi)
#pragma unroll
    for (int ni = 0; ni < 4; ++ni) {
      const long rb = row0 + mi * 16 + hi * 8;  // C/D layout: row = v + hi*8
      const long cc = col0 + ni * 16 + lo;      // col = lane&15
      if (epi == 0) {
#pragma unroll
        for (int v = 0; v < 8; ++v) {
          float val = acc[mi][ni][v];
          if (bias) val += bias[rb + v];
          outF[(rb + v) * (long)ldo + cc] = val;
        }
      } else {
        bf16x8 pk;
#pragma unroll
        for (int v = 0; v < 8; ++v) pk[v] = (bf16)(acc[mi][ni][v] * scale);
        *reinterpret_cast<bf16x8*>(outT + cc * (long)ldo + rb) = pk;
      }
    }
}

// ---------------------------------------------------------------------------
// 3x3 conv (pad=1) as 9-tap implicit GEMM.
// Wtap: [9][512][512] bf16 (A operand, row = co).
// xT:   [B][1024][1536] bf16, slice 0 holds x^T[n][ci] (B operand rows).
// zrow: 1024 zero bf16 -- OOB taps point here (no cndmask in the hot loop).
// transpose_out=1 -> out[b][n][512]; 0 -> out[b][512][1024].
// ---------------------------------------------------------------------------
__global__ __launch_bounds__(256) void conv3x3_kernel(
    const bf16* __restrict__ Wtap,
    const float* __restrict__ bias,
    const bf16* __restrict__ xT,
    const bf16* __restrict__ zrow,
    bf16* __restrict__ out,
    int transpose_out) {
  const int tid  = threadIdx.x;
  const int lane = tid & 31;
  const int wv   = tid >> 5;
  const int lo   = lane & 15;
  const int hi   = lane >> 4;
  const int mw   = wv >> 1;
  const int nw   = wv & 1;
  const int b    = blockIdx.z;

  const bf16* xTb = xT + (long)b * 1024 * 1536;
  const long row0 = (long)blockIdx.x * 128 + mw * 32;   // co
  const long col0 = (long)blockIdx.y * 128 + nw * 64;   // n = h*32+w

  f32x8 acc[2][4];
#pragma unroll
  for (int mi = 0; mi < 2; ++mi)
#pragma unroll
    for (int ni = 0; ni < 4; ++ni)
#pragma unroll
      for (int e = 0; e < 8; ++e) acc[mi][ni][e] = 0.0f;

  for (int tap = 0; tap < 9; ++tap) {
    const int kh = tap / 3, kw = tap % 3;
    const bf16* Atap = Wtap + (long)tap * 512 * 512;

    // Per-lane shifted source row; zero padding -> point at zero row.
    const bf16* br[4];
#pragma unroll
    for (int ni = 0; ni < 4; ++ni) {
      const int n  = (int)col0 + ni * 16 + lo;
      const int sh = (n >> 5) + kh - 1;
      const int sw = (n & 31) + kw - 1;
      const bool valid = ((unsigned)sh < 32u) && ((unsigned)sw < 32u);
      br[ni] = valid ? (xTb + (long)(sh * 32 + sw) * 1536) : zrow;
    }

    for (int kc = 0; kc < 512; kc += 32) {
      bf16x16 a0 = ld_frag_a(Atap, row0 + lo,      512, kc, hi);
      bf16x16 a1 = ld_frag_a(Atap, row0 + 16 + lo, 512, kc, hi);
      bf16x16 b0 = ld_frag_b_row(br[0], kc, hi);
      bf16x16 b1 = ld_frag_b_row(br[1], kc, hi);
      bf16x16 b2 = ld_frag_b_row(br[2], kc, hi);
      bf16x16 b3 = ld_frag_b_row(br[3], kc, hi);
      acc[0][0] = wmma_bf16(a0, b0, acc[0][0]);
      acc[0][1] = wmma_bf16(a0, b1, acc[0][1]);
      acc[0][2] = wmma_bf16(a0, b2, acc[0][2]);
      acc[0][3] = wmma_bf16(a0, b3, acc[0][3]);
      acc[1][0] = wmma_bf16(a1, b0, acc[1][0]);
      acc[1][1] = wmma_bf16(a1, b1, acc[1][1]);
      acc[1][2] = wmma_bf16(a1, b2, acc[1][2]);
      acc[1][3] = wmma_bf16(a1, b3, acc[1][3]);
    }
  }

#pragma unroll
  for (int mi = 0; mi < 2; ++mi)
#pragma unroll
    for (int ni = 0; ni < 4; ++ni) {
      const long rb = row0 + mi * 16 + hi * 8;
      const long cc = col0 + ni * 16 + lo;
      if (transpose_out) {
        bf16x8 pk;
#pragma unroll
        for (int v = 0; v < 8; ++v) pk[v] = (bf16)(acc[mi][ni][v] + bias[rb + v]);
        *reinterpret_cast<bf16x8*>(out + ((long)b * 1024 + cc) * 512 + rb) = pk;
      } else {
#pragma unroll
        for (int v = 0; v < 8; ++v)
          out[((long)b * 512 + rb + v) * 1024 + cc] =
              (bf16)(acc[mi][ni][v] + bias[rb + v]);
      }
    }
}

// ---------------------------------------------------------------------------
// Row softmax, f32 in -> bf16 out. One 256-thread block per row.
// trans=1 writes Out[col*ldoT + row] (transposed, for P^T).
// ---------------------------------------------------------------------------
__global__ __launch_bounds__(256) void softmax_row_kernel(
    const float* __restrict__ In, bf16* __restrict__ Out,
    int len, int trans, int ldoT) {
  __shared__ float red[256];
  const int tid = threadIdx.x;
  const float* row = In + (long)blockIdx.x * len;

  float m = -3.0e38f;
  for (int i = tid; i < len; i += 256) m = fmaxf(m, row[i]);
  red[tid] = m; __syncthreads();
  for (int s = 128; s > 0; s >>= 1) {
    if (tid < s) red[tid] = fmaxf(red[tid], red[tid + s]);
    __syncthreads();
  }
  const float rmax = red[0]; __syncthreads();

  float sum = 0.0f;
  for (int i = tid; i < len; i += 256) sum += __expf(row[i] - rmax);
  red[tid] = sum; __syncthreads();
  for (int s = 128; s > 0; s >>= 1) {
    if (tid < s) red[tid] += red[tid + s];
    __syncthreads();
  }
  const float inv = 1.0f / red[0];

  for (int i = tid; i < len; i += 256) {
    float e = __expf(row[i] - rmax) * inv;
    if (trans) Out[(long)i * ldoT + blockIdx.x] = (bf16)e;
    else       Out[(long)blockIdx.x * len + i]  = (bf16)e;
  }
}

// ---------------------------------------------------------------------------
// Packing kernels (f32 -> bf16, layout rearrangement).
// ---------------------------------------------------------------------------
__global__ __launch_bounds__(256) void pack_x_kernel(
    const float* __restrict__ x, bf16* __restrict__ xc, bf16* __restrict__ catT) {
  const long i = (long)blockIdx.x * 256 + threadIdx.x;   // [b][n][c], c fastest
  const int c = (int)(i & 511);
  const long t = i >> 9;
  const int n = (int)(t & 1023);
  const int b = (int)(t >> 10);
  const float v = x[(((long)b * 512 + c) << 10) + n];
  xc[(((long)b * 512 + c) << 10) + n]   = (bf16)v;       // [b][c][n]
  catT[((long)b * 1024 + n) * 1536 + c] = (bf16)v;       // [b][n][1536], slice 0
}

__global__ __launch_bounds__(256) void pack_w_kernel(
    const float* __restrict__ w, bf16* __restrict__ wt) {
  const long i = (long)blockIdx.x * 256 + threadIdx.x;   // [tap][co][ci]
  const int ci = (int)(i & 511);
  const long t = i >> 9;
  const int co  = (int)(t & 511);
  const int tap = (int)(t >> 9);
  wt[i] = (bf16)w[((long)co * 512 + ci) * 9 + tap];
}

__global__ __launch_bounds__(256) void pack_wf_kernel(
    const float* __restrict__ wf, bf16* __restrict__ wfb) {
  const long i = (long)blockIdx.x * 256 + threadIdx.x;   // [co][1536] already
  wfb[i] = (bf16)wf[i];
}

__global__ __launch_bounds__(256) void zero_row_kernel(bf16* __restrict__ z) {
  const int i = blockIdx.x * 256 + threadIdx.x;          // 1024 elements
  z[i] = (bf16)0.0f;
}

// ---------------------------------------------------------------------------
extern "C" void kernel_launch(void* const* d_in, const int* in_sizes, int n_in,
                              void* d_out, int out_size, void* d_ws, size_t ws_size,
                              hipStream_t stream) {
  const float* x     = (const float*)d_in[0];
  const float* w1    = (const float*)d_in[1];
  const float* b1    = (const float*)d_in[2];
  const float* w2    = (const float*)d_in[3];
  const float* b2    = (const float*)d_in[4];
  const float* w3    = (const float*)d_in[5];
  const float* b3    = (const float*)d_in[6];
  const float* alpha = (const float*)d_in[7];
  const float* beta  = (const float*)d_in[8];
  const float* wf    = (const float*)d_in[9];
  const float* bfb   = (const float*)d_in[10];
  float* out = (float*)d_out;

  char* ws = (char*)d_ws;
  size_t off = 0;
  auto carve = [&](size_t bytes) -> char* {
    char* p = ws + off;
    off += (bytes + 255) & ~(size_t)255;
    return p;
  };
  bf16*  xc   = (bf16*) carve((size_t)16 * 512 * 1024 * 2);   // x  [b][c][n]
  bf16*  catT = (bf16*) carve((size_t)16 * 1024 * 1536 * 2);  // [xT | a*spatT | b*chanT]
  bf16*  At   = (bf16*) carve((size_t)16 * 1024 * 512 * 2);   // conv1^T [b][n][c]
  bf16*  Bt   = (bf16*) carve((size_t)16 * 1024 * 512 * 2);   // conv2^T [b][n][c]
  bf16*  Dc   = (bf16*) carve((size_t)16 * 512 * 1024 * 2);   // conv3   [b][c][n]
  float* M1   = (float*)carve((size_t)1024 * 1024 * 4);       // per-batch logits
  bf16*  Msm1 = (bf16*) carve((size_t)1024 * 1024 * 2);       // softmax(M) [n][m]
  float* S1   = (float*)carve((size_t)512 * 512 * 4);         // per-batch Gram
  bf16*  PT1  = (bf16*) carve((size_t)512 * 512 * 2);         // softmax(S)^T [c][d]
  bf16*  w1b  = (bf16*) carve((size_t)9 * 512 * 512 * 2);
  bf16*  w2b  = (bf16*) carve((size_t)9 * 512 * 512 * 2);
  bf16*  w3b  = (bf16*) carve((size_t)9 * 512 * 512 * 2);
  bf16*  wfbp = (bf16*) carve((size_t)512 * 1536 * 2);
  bf16*  zrow = (bf16*) carve((size_t)1024 * 2);              // zero row for conv halo

  const dim3 blk(256);

  // ---- pack to bf16 -------------------------------------------------------
  pack_x_kernel <<<32768, blk, 0, stream>>>(x, xc, catT);
  pack_w_kernel <<<9216,  blk, 0, stream>>>(w1, w1b);
  pack_w_kernel <<<9216,  blk, 0, stream>>>(w2, w2b);
  pack_w_kernel <<<9216,  blk, 0, stream>>>(w3, w3b);
  pack_wf_kernel<<<3072,  blk, 0, stream>>>(wf, wfbp);
  zero_row_kernel<<<4,    blk, 0, stream>>>(zrow);

  // ---- 3x3 convs (implicit GEMM), block tile 128x128 ----------------------
  conv3x3_kernel<<<dim3(4, 8, 16), blk, 0, stream>>>(w1b, b1, catT, zrow, At, 1);
  conv3x3_kernel<<<dim3(4, 8, 16), blk, 0, stream>>>(w2b, b2, catT, zrow, Bt, 1);
  conv3x3_kernel<<<dim3(4, 8, 16), blk, 0, stream>>>(w3b, b3, catT, zrow, Dc, 0);

  // ---- spatial attention (per batch: M, softmax, spat) -------------------
  for (int b = 0; b < 16; ++b) {
    const bf16* Atb = At + (size_t)b * 1024 * 512;
    const bf16* Btb = Bt + (size_t)b * 1024 * 512;
    // M[n][m] = sum_c A^T[n][c] * B^T[m][c]
    gemm_nt_kernel<<<dim3(8, 8, 1), blk, 0, stream>>>(
        512, Atb, 512, 0, Btb, 512, 0, M1, nullptr, 1024, 0,
        nullptr, nullptr, /*epi=*/0);
    softmax_row_kernel<<<1024, blk, 0, stream>>>(M1, Msm1, 1024, 0, 0);
    // spat[c][n] = sum_m D[c][m] * Msm[n][m]; write alpha*spat^T into catT slice 1
    gemm_nt_kernel<<<dim3(4, 8, 1), blk, 0, stream>>>(
        1024, Dc + (size_t)b * 512 * 1024, 1024, 0, Msm1, 1024, 0,
        nullptr, catT + (size_t)b * 1024 * 1536 + 512, 1536, 0,
        alpha, nullptr, /*epi=*/1);
  }

  // ---- channel attention (per batch: S, softmax^T, chan) -----------------
  for (int b = 0; b < 16; ++b) {
    const bf16* xcb = xc + (size_t)b * 512 * 1024;
    // S[c][d] = sum_n x[c][n] * x[d][n]
    gemm_nt_kernel<<<dim3(4, 4, 1), blk, 0, stream>>>(
        1024, xcb, 1024, 0, xcb, 1024, 0, S1, nullptr, 512, 0,
        nullptr, nullptr, /*epi=*/0);
    softmax_row_kernel<<<512, blk, 0, stream>>>(S1, PT1, 512, 1, 512);
    // chan[c][n] = sum_d P^T[c][d] * x[d][n]; B operand = x^T rows in catT slice 0
    gemm_nt_kernel<<<dim3(4, 8, 1), blk, 0, stream>>>(
        512, PT1, 512, 0, catT + (size_t)b * 1024 * 1536, 1536, 0,
        nullptr, catT + (size_t)b * 1024 * 1536 + 1024, 1536, 0,
        beta, nullptr, /*epi=*/1);
  }

  // ---- fuse: out[co][n] = Wf[co][1536] . catT[n][1536] + bf --------------
  gemm_nt_kernel<<<dim3(4, 8, 16), blk, 0, stream>>>(
      1536, wfbp, 1536, 0, catT, 1536, (long)1024 * 1536,
      out, nullptr, 1024, (long)512 * 1024,
      nullptr, bfb, /*epi=*/0);
}
